// ForeCastModel_39238821216907
// MI455X (gfx1250) — matrix operand order repaired and checked by reading, not compiled, not run on hardware
//
#include <hip/hip_runtime.h>

// ---------------- problem constants ----------------
#define ZDIM   8
#define HDIM   91
#define WDIMX  180
#define CDIM   192
#define HEADS  6
#define DH     32
#define HPP    96
#define TYPEW  64     // NZ*NH
#define TOK    144
#define TOKP   160    // TOK padded to multiple of 32 for the PV K-loop
#define NWIN   960    // NW*TYPEW
#define NTOKW  (NWIN*TOK)        // 138240 window tokens
#define NTOK   (ZDIM*HDIM*WDIMX) // 131040 real tokens
#define BIASROWS 3312
#define KCH    192    // K chunk staged in LDS per iteration (24KB bf16 panel)

typedef __bf16 bf16;
typedef __attribute__((ext_vector_type(16))) __bf16 v16bf;
typedef __attribute__((ext_vector_type(8)))  float  v8f;
typedef __attribute__((ext_vector_type(4)))  unsigned int u32x4;
typedef __attribute__((ext_vector_type(4)))  int i32x4;
typedef __attribute__((ext_vector_type(8)))  int i32x8;

union AB32 { u32x4 u[2]; v16bf v; };

// ---- WMMA fragment loaders (CDNA5 wave32 layouts, CDNA5 ISA) ----
// A 16x32 bf16: lane<16 row=lane, K = {hf*8+e | e<8} U {16+hf*8+(e-8)}
__device__ __forceinline__ v16bf load_a_frag(const bf16* base, int ld) {
  const int lane = threadIdx.x & 31;
  const int r = lane & 15, hf = lane >> 4;
  const bf16* p = base + (size_t)r * ld + hf * 8;
  AB32 t;
  t.u[0] = *(const u32x4*)(p);
  t.u[1] = *(const u32x4*)(p + 16);
  return t.v;
}
// B 32x16 bf16 from Bt[n][k] (n-major): col=lane%16, K = {hf*8+e | e<8} U {16+hf*8+(e-8)}
__device__ __forceinline__ v16bf load_b_frag(const bf16* base, int ld) {
  const int lane = threadIdx.x & 31;
  const int r = lane & 15, hf = lane >> 4;
  const bf16* p = base + (size_t)r * ld + hf * 8;
  AB32 t;
  t.u[0] = *(const u32x4*)(p);
  t.u[1] = *(const u32x4*)(p + 16);
  return t.v;
}
__device__ __forceinline__ v8f wmma_bf16(v16bf a, v16bf b, v8f c) {
  return __builtin_amdgcn_wmma_f32_16x16x32_bf16(false, a, false, b, (short)0, c, false, false);
}

// ---- stage a 64 x KCH bf16 weight panel (rows ns*64.., cols kc..) into LDS ----
// Prefer the Tensor Data Mover (CDNA5 ISA): wave 0 issues one
// tensor_load_to_lds, waits on TENSORcnt, then the workgroup barriers.
__device__ __forceinline__ void stage_panel(bf16* panel, const bf16* __restrict__ Wt,
                                            int ns, int kc, int K) {
#if __has_builtin(__builtin_amdgcn_tensor_load_to_lds)
  if (threadIdx.x < 32) {
    const bf16* gsrc = Wt + (size_t)(ns * 64) * K + kc;
    unsigned long long ga = (unsigned long long)gsrc;
    unsigned int lds_off = (unsigned int)(unsigned long long)panel;  // LDS flat addr keeps byte offset in [31:0]
    // D# group 0: count=1 | lds_addr | global_addr[56:0] | type=2
    u32x4 g0;
    g0.x = 1u;
    g0.y = lds_off;
    g0.z = (unsigned int)(ga & 0xffffffffull);
    g0.w = (unsigned int)((ga >> 32) & 0x1ffffffull) | (2u << 30);
    // D# group 1: data_size=1 (2B); tensor dims = tile dims (no OOB);
    // tile_dim0=KCH, tile_dim1=64; tensor_dim0_stride = K (row stride in elems)
    i32x8 g1;
    g1[0] = (int)(1u << 16);                         // data_size=1
    g1[1] = (int)((KCH & 0xffff) << 16);             // tensor_dim0[15:0] @ bits 63:48
    g1[2] = (int)((KCH >> 16) | (64 << 16));         // tensor_dim0[31:16] | tensor_dim1[15:0]
    g1[3] = (int)((64 >> 16) | (KCH << 16));         // tensor_dim1[31:16] | tile_dim0
    g1[4] = 64;                                      // tile_dim1=64, tile_dim2=0 (2D)
    g1[5] = K;                                       // tensor_dim0_stride[31:0]
    g1[6] = 0;
    g1[7] = 0;
    i32x4 gz = {0, 0, 0, 0};
#if __clang_major__ >= 23
    i32x8 gz8 = {0, 0, 0, 0, 0, 0, 0, 0};
    __builtin_amdgcn_tensor_load_to_lds(g0, g1, gz, gz, gz8, 0);
#else
    __builtin_amdgcn_tensor_load_to_lds(g0, g1, gz, gz, 0);
#endif
    __builtin_amdgcn_s_wait_tensorcnt(0);
  }
#else
  // fallback: cooperative dword copy
  unsigned int* dst = (unsigned int*)panel;
  for (int i = threadIdx.x; i < 64 * (KCH / 2); i += 256) {
    int row = i / (KCH / 2), c2 = i % (KCH / 2);
    dst[i] = ((const unsigned int*)(Wt + (size_t)(ns * 64 + row) * K + kc))[c2];
  }
#endif
  __syncthreads();
}

// ---- shared GEMM body: block computes a 256(M) x 64(N) tile ----------------
// 8 waves x (2 M-tiles x 4 N-subtiles); B panel staged in LDS per K-chunk.
template <typename F>
__device__ __forceinline__ void gemm_panel(const bf16* __restrict__ A,
                                           const bf16* __restrict__ Wt,
                                           int Mtiles, int N, int K, F&& emit) {
  __shared__ __align__(16) bf16 panel[64 * KCH];
  const int NS = N >> 6;
  const int ns = blockIdx.x % NS;
  const int mtBase = (blockIdx.x / NS) * 16;
  const int wave = threadIdx.x >> 5;
  const int lane = threadIdx.x & 31;
  const int r = lane & 15, hf = lane >> 4;
  const int mt0 = mtBase + wave * 2;
  const int mt1 = mt0 + 1;
  const bool ok0 = mt0 < Mtiles, ok1 = mt1 < Mtiles;
  const bf16* A0 = A + (size_t)(ok0 ? mt0 : 0) * 16 * K;
  const bf16* A1 = A + (size_t)(ok1 ? mt1 : (ok0 ? mt0 : 0)) * 16 * K;

  v8f acc0[4], acc1[4];
  v8f zz = {0, 0, 0, 0, 0, 0, 0, 0};
#pragma unroll
  for (int s = 0; s < 4; ++s) { acc0[s] = zz; acc1[s] = zz; }

  for (int kc = 0; kc < K; kc += KCH) {
    if (kc) __syncthreads();   // panel reuse: wait for readers of previous chunk
    stage_panel(panel, Wt, ns, kc, K);
#pragma unroll
    for (int kb = 0; kb < KCH; kb += 32) {
      v16bf a0 = load_a_frag(A0 + kc + kb, K);
      v16bf a1 = load_a_frag(A1 + kc + kb, K);
#pragma unroll
      for (int s = 0; s < 4; ++s) {
        v16bf b = load_b_frag(panel + (s * 16) * KCH + kb, KCH);
        acc0[s] = wmma_bf16(a0, b, acc0[s]);
        acc1[s] = wmma_bf16(a1, b, acc1[s]);
      }
    }
  }
#pragma unroll
  for (int s = 0; s < 4; ++s) {
    int n = ns * 64 + s * 16 + r;
#pragma unroll
    for (int v = 0; v < 8; ++v) {
      if (ok0) emit(mt0 * 16 + v + hf * 8, n, acc0[s][v]);
      if (ok1) emit(mt1 * 16 + v + hf * 8, n, acc1[s][v]);
    }
  }
}

// ---------------- weight cast + transpose: W[K][N] f32 -> Wt[N][K] bf16 ----
__global__ void cast_wt_kernel(const float* __restrict__ w, bf16* __restrict__ wt,
                               int K, int N) {
  int idx = blockIdx.x * blockDim.x + threadIdx.x;
  if (idx >= K * N) return;
  int n = idx / K, k = idx % K;
  wt[idx] = (bf16)w[(size_t)k * N + n];
}

// ---------------- window gather (+pad +roll) to bf16 [NWIN][TOK][CDIM] ------
__global__ void gather_win_kernel(const float* __restrict__ x, bf16* __restrict__ xw,
                                  int roll) {
  size_t idx = (size_t)blockIdx.x * blockDim.x + threadIdx.x;
  if (idx >= (size_t)NWIN * TOK * 96) return;
  int dp = (int)(idx % 96);
  int t  = (int)((idx / 96) % TOK);
  int win = (int)(idx / (96 * TOK));
  int p = win & 63, nw = win >> 6;
  int nz = p >> 4, nh = p & 15;
  int wz = t / 72, rem = t % 72, wh = rem / 12, ww = rem % 12;
  int z = nz * 2 + wz, h = nh * 6 + wh, w = nw * 12 + ww;
  if (roll) { z = (z + 1) & 7; h = (h + 3) % HPP; w = (w + 6) % WDIMX; }
  float v0 = 0.f, v1 = 0.f;
  if (h < HDIM) {
    const float* s = x + (((size_t)z * HDIM + h) * WDIMX + w) * CDIM + dp * 2;
    v0 = s[0]; v1 = s[1];
  }
  bf16* o = xw + ((size_t)win * TOK + t) * CDIM + dp * 2;
  o[0] = (bf16)v0; o[1] = (bf16)v1;
}

// ---------------- zero the K-padding columns of v^T [wh][32][TOKP] ---------
__global__ void pad_vt_kernel(bf16* __restrict__ vt) {
  int idx = blockIdx.x * blockDim.x + threadIdx.x;
  if (idx >= NWIN * HEADS * DH * 16) return;
  int c = idx & 15, dr = idx >> 4;
  vt[(size_t)dr * TOKP + TOK + c] = (bf16)0.f;
}

// ---------------- QKV GEMM: XW @ Wt -> q (scaled), k, v^T (bf16) -----------
__global__ __launch_bounds__(256) void gemm_qkv_kernel(
    const bf16* __restrict__ A, const bf16* __restrict__ Wt,
    const float* __restrict__ bias, bf16* __restrict__ qo,
    bf16* __restrict__ ko, bf16* __restrict__ vto) {
  gemm_panel(A, Wt, NTOKW / 16, 576, CDIM, [&](int m, int n, float val) {
    val += bias[n];
    int which = n / CDIM, remn = n % CDIM, head = remn >> 5, d = remn & 31;
    int win = m / TOK, tok = m % TOK;
    size_t wh = (size_t)win * HEADS + head;
    if (which == 0)      qo[(wh * TOK + tok) * DH + d] = (bf16)(val * 0.17677669529663687f);
    else if (which == 1) ko[(wh * TOK + tok) * DH + d] = (bf16)val;
    else                 vto[(wh * DH + d) * TOKP + tok] = (bf16)val;
  });
}

// ---------------- generic GEMM -> f32 out (proj, mlp2) ---------------------
__global__ __launch_bounds__(256) void gemm_f32_kernel(
    const bf16* __restrict__ A, const bf16* __restrict__ Wt,
    const float* __restrict__ bias, float* __restrict__ out,
    int Mtiles, int N, int K) {
  gemm_panel(A, Wt, Mtiles, N, K, [&](int m, int n, float val) {
    out[(size_t)m * N + n] = val + bias[n];
  });
}

// ---------------- GEMM + exact GELU -> bf16 out (mlp1) ---------------------
__global__ __launch_bounds__(256) void gemm_gelu_kernel(
    const bf16* __restrict__ A, const bf16* __restrict__ Wt,
    const float* __restrict__ bias, bf16* __restrict__ out,
    int Mtiles, int N, int K) {
  gemm_panel(A, Wt, Mtiles, N, K, [&](int m, int n, float val) {
    float t = val + bias[n];
    float g = 0.5f * t * (1.f + erff(t * 0.70710678118654752f));
    out[(size_t)m * N + n] = (bf16)g;
  });
}

// ---------------- attention: one block per (window, head) ------------------
__global__ __launch_bounds__(288) void attn_kernel(
    const bf16* __restrict__ qg, const bf16* __restrict__ kg,
    const bf16* __restrict__ vtg, const float* __restrict__ btab,
    bf16* __restrict__ og, int roll) {
  const int wh = blockIdx.x;                 // win*HEADS + head
  const int win = wh / HEADS, head = wh % HEADS;
  const int p = win & 63;                    // window type (nz*16+nh)
  const int wave = threadIdx.x >> 5;
  const int lane = threadIdx.x & 31;
  const int r = lane & 15, hf = lane >> 4;

  __shared__ __align__(16) bf16 P[9][16][TOKP];

  const bf16* qb = qg + (size_t)wh * (TOK * DH);
  const bf16* kb = kg + (size_t)wh * (TOK * DH);
  const bf16* vb = vtg + (size_t)wh * (DH * TOKP);

  // ---- S = q @ k^T  (9 tiles of 16x16, K = dh = 32)
  v8f s[9];
  {
    v16bf a = load_a_frag(qb + (size_t)(wave * 16) * DH, DH);
    v8f z = {0, 0, 0, 0, 0, 0, 0, 0};
#pragma unroll
    for (int tj = 0; tj < 9; ++tj) {
      v16bf b = load_b_frag(kb + (size_t)(tj * 16) * DH, DH);
      s[tj] = wmma_bf16(a, b, z);
    }
  }

  // ---- per-row invariants for bias index & shift-mask region
  int zi[8], hi[8], wi[8], regi[8];
#pragma unroll
  for (int v = 0; v < 8; ++v) {
    int i = wave * 16 + v + hf * 8;
    zi[v] = i / 72; int rem = i % 72; hi[v] = rem / 12; wi[v] = rem % 12;
    int zz = (p >> 4) * 2 + zi[v];
    int hhp = (p & 15) * 6 + hi[v];
    int zr = (zz < 6) ? 0 : (zz == 6 ? 1 : 2);
    int hr = (hhp < 90) ? 0 : (hhp <= 92 ? 1 : 2);
    regi[v] = zr * 3 + hr;
  }
  // ---- add earth bias (+ mask when rolled)
#pragma unroll
  for (int tj = 0; tj < 9; ++tj) {
    int j = tj * 16 + r;
    int zj = j / 72; int rem = j % 72; int hj = rem / 12; int wj = rem % 12;
    int zz = (p >> 4) * 2 + zj;
    int hhp = (p & 15) * 6 + hj;
    int zr = (zz < 6) ? 0 : (zz == 6 ? 1 : 2);
    int hr = (hhp < 90) ? 0 : (hhp <= 92 ? 1 : 2);
    int regj = zr * 3 + hr;
#pragma unroll
    for (int v = 0; v < 8; ++v) {
      int bidx = (zi[v] + 2 * zj) * 828 + (hi[v] + 6 * hj) * 23 + (wi[v] - wj + 11);
      float bv = btab[((size_t)bidx * TYPEW + p) * HEADS + head];
      float val = s[tj][v] + bv;
      if (roll && (regi[v] != regj)) val -= 100.f;
      s[tj][v] = val;
    }
  }

  // ---- row softmax (rows live across 16 lanes per half-wave in C layout)
#pragma unroll
  for (int v = 0; v < 8; ++v) {
    float mx = s[0][v];
#pragma unroll
    for (int tj = 1; tj < 9; ++tj) mx = fmaxf(mx, s[tj][v]);
    for (int d = 1; d < 16; d <<= 1) mx = fmaxf(mx, __shfl_xor(mx, d, 32));
    float sum = 0.f;
#pragma unroll
    for (int tj = 0; tj < 9; ++tj) { float e = __expf(s[tj][v] - mx); s[tj][v] = e; sum += e; }
    for (int d = 1; d < 16; d <<= 1) sum += __shfl_xor(sum, d, 32);
    float inv = 1.f / sum;
#pragma unroll
    for (int tj = 0; tj < 9; ++tj) s[tj][v] *= inv;
  }

  // ---- P -> LDS (bf16, K-padded with zeros)
#pragma unroll
  for (int tj = 0; tj < 9; ++tj)
#pragma unroll
    for (int v = 0; v < 8; ++v)
      P[wave][v + hf * 8][tj * 16 + r] = (bf16)s[tj][v];
#pragma unroll
  for (int v = 0; v < 8; ++v)
    P[wave][v + hf * 8][TOK + r] = (bf16)0.f;
  __syncthreads();

  // ---- O = P @ V (K = TOKP = 160, v stored transposed [dh][TOKP])
#pragma unroll
  for (int db = 0; db < 2; ++db) {
    v8f o = {0, 0, 0, 0, 0, 0, 0, 0};
#pragma unroll
    for (int kb2 = 0; kb2 < TOKP / 32; ++kb2) {
      v16bf a = load_a_frag(&P[wave][0][0] + kb2 * 32, TOKP);
      v16bf b = load_b_frag(vb + (size_t)(db * 16) * TOKP + kb2 * 32, TOKP);
      o = wmma_bf16(a, b, o);
    }
#pragma unroll
    for (int v = 0; v < 8; ++v) {
      int tok = wave * 16 + v + hf * 8;
      int ch = head * DH + db * 16 + r;
      og[((size_t)win * TOK + tok) * CDIM + ch] = (bf16)o[v];
    }
  }
}

// ---------------- un-window + un-roll + crop + residual + LN1 --------------
__global__ void epilogue1_kernel(const float* __restrict__ proj,
                                 const float* __restrict__ xprev,
                                 const float* __restrict__ g1, const float* __restrict__ b1,
                                 float* __restrict__ x1, bf16* __restrict__ x1bf, int roll) {
  int t = blockIdx.x * blockDim.x + threadIdx.x;
  if (t >= NTOK) return;
  int z = t / (HDIM * WDIMX);
  int rem = t % (HDIM * WDIMX);
  int h = rem / WDIMX, w = rem % WDIMX;
  int zy = z, hy = h, wy = w;
  if (roll) { zy = (z + 7) & 7; hy = (h + 93) % HPP; wy = (w + 174) % WDIMX; }
  int nz = zy >> 1, wz = zy & 1;
  int nh = hy / 6, whh = hy % 6;
  int nw = wy / 12, ww = wy % 12;
  int win = nw * 64 + nz * 16 + nh;
  int tok = wz * 72 + whh * 12 + ww;
  const float* prow = proj + ((size_t)win * TOK + tok) * CDIM;
  float mean = 0.f;
  for (int d = 0; d < CDIM; ++d) mean += prow[d];
  mean *= (1.f / CDIM);
  float var = 0.f;
  for (int d = 0; d < CDIM; ++d) { float df = prow[d] - mean; var += df * df; }
  var *= (1.f / CDIM);
  float rs = rsqrtf(var + 1e-5f);
  const float* srow = xprev + (size_t)t * CDIM;
  float* orow = x1 + (size_t)t * CDIM;
  bf16* brow = x1bf + (size_t)t * CDIM;
  for (int d = 0; d < CDIM; ++d) {
    float val = srow[d] + (prow[d] - mean) * rs * g1[d] + b1[d];
    orow[d] = val;
    brow[d] = (bf16)val;
  }
}

// ---------------- residual + LN2 (per chunk) -------------------------------
__global__ void ln2_kernel(const float* __restrict__ h2, const float* __restrict__ x1,
                           const float* __restrict__ g, const float* __restrict__ b,
                           float* __restrict__ out, int mcount) {
  int t = blockIdx.x * blockDim.x + threadIdx.x;
  if (t >= mcount) return;
  const float* hr = h2 + (size_t)t * CDIM;
  float mean = 0.f;
  for (int d = 0; d < CDIM; ++d) mean += hr[d];
  mean *= (1.f / CDIM);
  float var = 0.f;
  for (int d = 0; d < CDIM; ++d) { float df = hr[d] - mean; var += df * df; }
  var *= (1.f / CDIM);
  float rs = rsqrtf(var + 1e-5f);
  const float* xr = x1 + (size_t)t * CDIM;
  float* orow = out + (size_t)t * CDIM;
  for (int d = 0; d < CDIM; ++d)
    orow[d] = xr[d] + (hr[d] - mean) * rs * g[d] + b[d];
}

// ============================ host driver ==================================
extern "C" void kernel_launch(void* const* d_in, const int* in_sizes, int n_in,
                              void* d_out, int out_size, void* d_ws, size_t ws_size,
                              hipStream_t stream) {
  (void)in_sizes; (void)n_in; (void)out_size; (void)ws_size;
  const float* x0     = (const float*)d_in[0];
  const float* qkv_w  = (const float*)d_in[1];
  const float* qkv_b  = (const float*)d_in[2];
  const float* proj_w = (const float*)d_in[3];
  const float* proj_b = (const float*)d_in[4];
  const float* btab   = (const float*)d_in[5];
  const float* n1g    = (const float*)d_in[6];
  const float* n1b    = (const float*)d_in[7];
  const float* n2g    = (const float*)d_in[8];
  const float* n2b    = (const float*)d_in[9];
  const float* w1     = (const float*)d_in[10];
  const float* b1m    = (const float*)d_in[11];
  const float* w2     = (const float*)d_in[12];
  const float* b2m    = (const float*)d_in[13];
  float* out = (float*)d_out;
  char* ws = (char*)d_ws;

  // ---- workspace layout (all 256B aligned); time-sliced aliasing
  const size_t OFF_WTQ = 0;                         // 576*192 bf16
  const size_t OFF_WTP = OFF_WTQ + 221184;          // 192*192 bf16
  const size_t OFF_W1T = OFF_WTP + 73728;           // 768*192 bf16
  const size_t OFF_W2T = OFF_W1T + 294912;          // 192*768 bf16
  const size_t OFF_XW  = OFF_W2T + 294912;          // 53.08MB: XW -> attnO -> mlp hidden chunk
  const size_t OFF_Q   = OFF_XW + 53084160;         // 53.08MB: q -> (q+k = proj f32) -> h2 chunk
  const size_t OFF_K   = OFF_Q + 53084160;          // 53.08MB: k
  const size_t OFF_VT  = OFF_K + 53084160;          // 58.98MB: v^T -> x1 bf16
  const size_t OFF_X1  = OFF_VT + 58982400;         // 100.64MB: x after attn residual+LN1

  bf16* WTQ = (bf16*)(ws + OFF_WTQ);
  bf16* WTP = (bf16*)(ws + OFF_WTP);
  bf16* W1T = (bf16*)(ws + OFF_W1T);
  bf16* W2T = (bf16*)(ws + OFF_W2T);
  bf16* XW  = (bf16*)(ws + OFF_XW);
  bf16* Qb  = (bf16*)(ws + OFF_Q);
  bf16* Kb  = (bf16*)(ws + OFF_K);
  bf16* VT  = (bf16*)(ws + OFF_VT);
  float* X1 = (float*)(ws + OFF_X1);
  bf16* ATTNO = XW;                   // alias (XW dead after QKV)
  float* PROJ = (float*)(ws + OFF_Q); // alias q+k (dead after attention)
  bf16* X1BF  = (bf16*)(ws + OFF_VT); // alias v^T (dead after attention)
  bf16* HCH   = XW;                   // alias: mlp hidden chunk (<=50.4MB)
  float* H2CH = (float*)(ws + OFF_Q); // alias: mlp out chunk (<=25.2MB)

  const int TPB = 256;
  for (int dep = 0; dep < 2; ++dep) {
    const int roll = dep;  // block 1 is shifted
    const float* xcur = (dep == 0) ? x0 : out;

    // weights -> bf16 transposed [N][K]
    cast_wt_kernel<<<(192*576 + TPB-1)/TPB, TPB, 0, stream>>>(qkv_w + (size_t)dep*192*576, WTQ, 192, 576);
    cast_wt_kernel<<<(192*192 + TPB-1)/TPB, TPB, 0, stream>>>(proj_w + (size_t)dep*192*192, WTP, 192, 192);
    cast_wt_kernel<<<(192*768 + TPB-1)/TPB, TPB, 0, stream>>>(w1 + (size_t)dep*192*768, W1T, 192, 768);
    cast_wt_kernel<<<(768*192 + TPB-1)/TPB, TPB, 0, stream>>>(w2 + (size_t)dep*768*192, W2T, 768, 192);

    // window partition (+roll, +pad) -> bf16
    {
      size_t n = (size_t)NWIN * TOK * 96;
      gather_win_kernel<<<(unsigned)((n + TPB-1)/TPB), TPB, 0, stream>>>(xcur, XW, roll);
    }
    pad_vt_kernel<<<(NWIN*HEADS*DH*16 + TPB-1)/TPB, TPB, 0, stream>>>(VT);

    // QKV GEMM -> q,k,v^T   (540 M-blocks x 9 N-strips)
    gemm_qkv_kernel<<<(NTOKW/256) * 9, TPB, 0, stream>>>(XW, WTQ, qkv_b + dep*576, Qb, Kb, VT);

    // attention
    attn_kernel<<<NWIN*HEADS, 288, 0, stream>>>(
        Qb, Kb, VT, btab + (size_t)dep*BIASROWS*TYPEW*HEADS, ATTNO, roll);

    // proj GEMM -> f32      (540 M-blocks x 3 N-strips)
    gemm_f32_kernel<<<(NTOKW/256) * 3, TPB, 0, stream>>>(ATTNO, WTP, proj_b + dep*CDIM,
                                                         PROJ, NTOKW/16, CDIM, CDIM);

    // un-window + residual + LN1 -> X1 (f32) + X1BF (bf16)
    epilogue1_kernel<<<(NTOK + TPB-1)/TPB, TPB, 0, stream>>>(
        PROJ, xcur, n1g + dep*CDIM, n1b + dep*CDIM, X1, X1BF, roll);

    // MLP chunked over tokens
    const int CH = 32768;
    for (int base = 0; base < NTOK; base += CH) {
      int mc = (NTOK - base) < CH ? (NTOK - base) : CH;  // 32768 / 32736, both %16==0
      int mt = mc / 16;
      int mb = (mt + 15) / 16;  // M-blocks of 256 rows (tail guarded in-kernel)
      gemm_gelu_kernel<<<mb * 12, TPB, 0, stream>>>(
          X1BF + (size_t)base*CDIM, W1T, b1m + dep*768, HCH, mt, 768, CDIM);
      gemm_f32_kernel<<<mb * 3, TPB, 0, stream>>>(
          HCH, W2T, b2m + dep*CDIM, H2CH, mt, CDIM, 768);
      ln2_kernel<<<(mc + TPB-1)/TPB, TPB, 0, stream>>>(
          H2CH, X1 + (size_t)base*CDIM, n2g + dep*CDIM, n2b + dep*CDIM,
          out + (size_t)base*CDIM, mc);
    }
  }
}
